// Longformer_24696061952653
// MI455X (gfx1250) — matrix operand order repaired and checked by reading, not compiled, not run on hardware
//
#include <hip/hip_runtime.h>
#include <hip/hip_bf16.h>

// ---------------------------------------------------------------------------
// Longformer forward for MI455X (gfx1250), wave32 + WMMA bf16.
// - All GEMMs and both attention matmuls: v_wmma_f32_16x16x32_bf16, f32 accum.
// - GEMM stages tiles with GLOBAL_LOAD_ASYNC_TO_LDS_B128 (ASYNCcnt) and
//   double-buffers LDS so DMA of chunk k+1 overlaps WMMA of chunk k.
// - Weights pre-transposed to [N][K] bf16 so B-fragments are two contiguous 16B
//   loads per lane per the CDNA5 WMMA VGPR layouts.
// ---------------------------------------------------------------------------

typedef unsigned short u16;
typedef unsigned int   u32;
typedef unsigned long long u64;

typedef __attribute__((ext_vector_type(16))) __bf16 v16bf;
typedef __attribute__((ext_vector_type(8)))  float  v8f;
typedef int v4i_vec __attribute__((__vector_size__(16)));   // matches builtin's V4i

#define BB    2
#define SS    2048
#define NFEAT 16
#define DD    768
#define HH    12
#define DH    64
#define LLAY  4
#define FF    3072
#define WINW  256
#define NCLS  15
#define BS    (BB * SS)     // 4096 tokens
#define QKVN  (3 * DD)      // 2304 fused QKV columns

union FragBF { v16bf v; uint4 q[2]; };
union FragC  { v8f   v; float f[8]; };

static __device__ __forceinline__ u16 f2bf(float f) {
  u32 u = __float_as_uint(f);
  u32 r = (u + 0x7FFFu + ((u >> 16) & 1u)) >> 16;   // round-to-nearest-even
  return (u16)r;
}

static __device__ __forceinline__ v8f wmma_bf16(v16bf a, v16bf b, v8f c) {
  // 8 args: (neg_a, A, neg_b, B, c_mod, C, reuse_a, reuse_b)
  return __builtin_amdgcn_wmma_f32_16x16x32_bf16(false, a, false, b, (short)0, c,
                                                 false, false);
}

// ---- async global->LDS copy (CDNA5 DMA path, ASYNCcnt tracked) ------------
#if defined(__has_builtin)
#if __has_builtin(__builtin_amdgcn_global_load_async_to_lds_b128)
#define HAVE_ASYNC_LDS 1
#endif
#endif

static __device__ __forceinline__ void async_cp16(void* lds, const void* g) {
#ifdef HAVE_ASYNC_LDS
  // signature: (v4i addrspace(1)*, v4i addrspace(3)*, imm offset, imm cpol)
  __builtin_amdgcn_global_load_async_to_lds_b128(
      (__attribute__((address_space(1))) v4i_vec*)(u64)(uintptr_t)g,
      (__attribute__((address_space(3))) v4i_vec*)(u32)(uintptr_t)lds,
      0, 0);
#else
  *(uint4*)lds = *(const uint4*)g;     // sync fallback
#endif
}
// Waits assemble on gfx1250 regardless; with the sync fallback ASYNCcnt is 0
// so they are no-ops.
#define ASYNC_WAIT_6() asm volatile("s_wait_asynccnt 0x6" ::: "memory")
#define ASYNC_WAIT_0() asm volatile("s_wait_asynccnt 0x0" ::: "memory")

// ---------------------------------------------------------------------------
// Weight prep: f32 W[K][N]  ->  bf16 WT[N][K]  (coalesced writes)
// ---------------------------------------------------------------------------
__global__ void k_transpose_bf16(const float* __restrict__ w, u16* __restrict__ wt,
                                 int K, int N) {
  int idx = blockIdx.x * blockDim.x + threadIdx.x;
  if (idx >= K * N) return;
  int n = idx / K;
  int k = idx - n * K;
  wt[idx] = f2bf(w[(size_t)k * N + n]);
}

// concat three 768-float bias vectors into one 2304-float vector
__global__ void k_concat3(const float* __restrict__ a, const float* __restrict__ b,
                          const float* __restrict__ c, float* __restrict__ o) {
  int i = blockIdx.x * blockDim.x + threadIdx.x;
  if (i < DD) o[i] = a[i];
  else if (i < 2 * DD) o[i] = b[i - DD];
  else if (i < 3 * DD) o[i] = c[i - 2 * DD];
}

// ---------------------------------------------------------------------------
// Embedding: e = x @ in_w + in_b + pos_emb[s+2] + tok_emb   (pre-LN, f32)
// ---------------------------------------------------------------------------
__global__ void k_embed(const float* __restrict__ x, const float* __restrict__ in_w,
                        const float* __restrict__ in_b, const float* __restrict__ pos_emb,
                        const float* __restrict__ tok_emb, float* __restrict__ outp) {
  int idx = blockIdx.x * blockDim.x + threadIdx.x;
  if (idx >= BS * DD) return;
  int t = idx / DD;
  int d = idx - t * DD;
  int s = t & (SS - 1);
  float acc = in_b[d] + tok_emb[d] + pos_emb[(size_t)(s + 2) * DD + d];
  const float* xr = x + (size_t)t * NFEAT;
#pragma unroll
  for (int kk = 0; kk < NFEAT; ++kk) acc += xr[kk] * in_w[(size_t)kk * DD + d];
  outp[idx] = acc;
}

// ---------------------------------------------------------------------------
// LayerNorm with optional residual: y = LN(inp (+ resid)); writes f32 + bf16
// ---------------------------------------------------------------------------
__global__ __launch_bounds__(256) void k_ln(const float* __restrict__ inp,
                                            const float* __restrict__ resid,
                                            const float* __restrict__ g,
                                            const float* __restrict__ bta,
                                            float* __restrict__ hout,
                                            u16* __restrict__ hbf) {
  __shared__ float s1[256], s2[256];
  const int row = blockIdx.x, tid = threadIdx.x;
  float vals[3], lsum = 0.f, lsq = 0.f;
#pragma unroll
  for (int i = 0; i < 3; ++i) {
    int d = tid + i * 256;
    float vv = inp[(size_t)row * DD + d];
    if (resid) vv += resid[(size_t)row * DD + d];
    vals[i] = vv; lsum += vv; lsq += vv * vv;
  }
  s1[tid] = lsum; s2[tid] = lsq;
  __syncthreads();
  for (int off = 128; off > 0; off >>= 1) {
    if (tid < off) { s1[tid] += s1[tid + off]; s2[tid] += s2[tid + off]; }
    __syncthreads();
  }
  const float mean = s1[0] * (1.0f / DD);
  const float var  = s2[0] * (1.0f / DD) - mean * mean;
  const float rs   = rsqrtf(var + 1e-12f);
#pragma unroll
  for (int i = 0; i < 3; ++i) {
    int d = tid + i * 256;
    float y = (vals[i] - mean) * rs * g[d] + bta[d];
    hout[(size_t)row * DD + d] = y;
    hbf[(size_t)row * DD + d]  = f2bf(y);
  }
}

// ---------------------------------------------------------------------------
// Tiled WMMA GEMM: C[M,N] = A[M,K](bf16) @ WT[N,K](bf16) + bias[N]
// 4 waves/block, 64(M)x128(N) block tile, 32x64 per wave (8 WMMA / K-chunk).
// K-chunks of 32 staged in double-buffered LDS via async global->LDS DMA.
// epi: 0 = bf16 store, 1 = exact GELU -> bf16 store, 2 = f32 store
// ---------------------------------------------------------------------------
__global__ __launch_bounds__(128) void k_gemm(const u16* __restrict__ A,
                                              const u16* __restrict__ WT,
                                              const float* __restrict__ bias,
                                              void* __restrict__ Cout,
                                              int M, int N, int K, int epi) {
  __shared__ u16 As[2][64 * 40];    // padded stride 40 elems keeps 16B align
  __shared__ u16 Bs[2][128 * 40];
  const int tid  = threadIdx.x;
  const int lane = tid & 31, wave = tid >> 5;
  const int lm = lane & 15, lg = lane >> 4;
  const int m0 = blockIdx.y * 64, n0 = blockIdx.x * 128;
  const int wm = (wave >> 1) * 32, wn = (wave & 1) * 64;
  (void)M;

  FragC acc[2][4];
#pragma unroll
  for (int i = 0; i < 2; ++i)
#pragma unroll
    for (int j = 0; j < 4; ++j)
#pragma unroll
      for (int r = 0; r < 8; ++r) acc[i][j].f[r] = 0.f;

  // issue one K-chunk of async DMA into LDS buffer `buf`
  auto stage = [&](int buf, int k0) {
#pragma unroll
    for (int jj = 0; jj < 2; ++jj) {            // A: 64 rows x 32 cols
      int i = tid * 2 + jj;
      int row = i >> 2, c4 = (i & 3) * 8;
      async_cp16(&As[buf][row * 40 + c4], A + (size_t)(m0 + row) * K + k0 + c4);
    }
#pragma unroll
    for (int jj = 0; jj < 4; ++jj) {            // B: 128 rows x 32 cols
      int i = tid * 4 + jj;
      int row = i >> 2, c4 = (i & 3) * 8;
      async_cp16(&Bs[buf][row * 40 + c4], WT + (size_t)(n0 + row) * K + k0 + c4);
    }
  };

  const int nk = K >> 5;
  stage(0, 0);
  for (int kc = 0; kc < nk; ++kc) {
    const int buf = kc & 1;
    if (kc + 1 < nk) {
      stage(buf ^ 1, (kc + 1) << 5);   // overlap next chunk's DMA with compute
      ASYNC_WAIT_6();                  // oldest 6 (current buf) complete
    } else {
      ASYNC_WAIT_0();
    }
    __syncthreads();

    FragBF af[2], bfr[4];
#pragma unroll
    for (int t = 0; t < 2; ++t) {      // A: m=lane&15, kb=(lane>>4)*8
      int ar = wm + t * 16 + lm;
      af[t].q[0] = *(const uint4*)(&As[buf][ar * 40 + lg * 8]);
      af[t].q[1] = *(const uint4*)(&As[buf][ar * 40 + lg * 8 + 16]);
    }
#pragma unroll
    for (int t = 0; t < 4; ++t) {      // B: n=lane&15, kb=(lane>>4)*8
      int br = wn + t * 16 + lm;
      bfr[t].q[0] = *(const uint4*)(&Bs[buf][br * 40 + lg * 8]);
      bfr[t].q[1] = *(const uint4*)(&Bs[buf][br * 40 + lg * 8 + 16]);
    }
#pragma unroll
    for (int i = 0; i < 2; ++i)
#pragma unroll
      for (int j = 0; j < 4; ++j)
        acc[i][j].v = wmma_bf16(af[i].v, bfr[j].v, acc[i][j].v);
    __syncthreads();
  }

  // epilogue: C layout lane=(n=lane&15, m=(lane>>4)*8 + r)
#pragma unroll
  for (int i = 0; i < 2; ++i)
#pragma unroll
    for (int j = 0; j < 4; ++j) {
      int n  = n0 + wn + j * 16 + lm;
      int mb = m0 + wm + i * 16 + lg * 8;
      float bvv = bias[n];
#pragma unroll
      for (int r = 0; r < 8; ++r) {
        float valf = acc[i][j].f[r] + bvv;
        if (epi == 1) valf = 0.5f * valf * (1.0f + erff(valf * 0.70710678118654752f));
        size_t off = (size_t)(mb + r) * N + n;
        if (epi == 2) ((float*)Cout)[off] = valf;
        else          ((u16*)Cout)[off]   = f2bf(valf);
      }
    }
}

// ---------------------------------------------------------------------------
// Sliding-window attention, one wave per (b, h, 16-query tile).
// q/k/v live in the fused QKV buffer with token stride qs (=2304).
// ---------------------------------------------------------------------------
__global__ __launch_bounds__(32) void k_attn(const u16* __restrict__ qg_,
                                             const u16* __restrict__ kg_,
                                             const u16* __restrict__ vg_,
                                             const int* __restrict__ amask,
                                             u16* __restrict__ outb, int qs) {
  __shared__ float sc[16][528];   // masked scores, keys q0-256 .. q0+271
  __shared__ u16   pr[16][544];   // exp(s - max) bf16, zero padded to 544
  __shared__ u16   vt[64][40];    // V chunk transposed: [dh][key], padded
  __shared__ float invs[16];      // 1 / rowsum
  const int lane = threadIdx.x;
  const int lm = lane & 15, lg = lane >> 4;
  const int q0 = blockIdx.x * 16;
  const int hh = blockIdx.y;
  const int bb = blockIdx.z;
  const size_t tokbase = (size_t)bb * SS;

  FragBF aq0, aq1;    // Q fragments, dh chunks 0..31 / 32..63
  {
    const u16* qp = qg_ + (tokbase + q0 + lm) * qs + hh * DH + lg * 8;
    aq0.q[0] = *(const uint4*)(qp + 0);
    aq0.q[1] = *(const uint4*)(qp + 16);
    aq1.q[0] = *(const uint4*)(qp + 32);
    aq1.q[1] = *(const uint4*)(qp + 48);
  }

  // ---- phase 1: scores = (Q K^T) * 1/sqrt(64), band + validity masked ----
  for (int kt = 0; kt < 33; ++kt) {
    const int kstart = q0 - WINW + kt * 16;
    const int kg = kstart + lm;              // this lane's key (B-frag n=lm)
    FragBF bk0, bk1;
    if (kg >= 0 && kg < SS) {
      const u16* kp = kg_ + (tokbase + kg) * qs + hh * DH + lg * 8;
      bk0.q[0] = *(const uint4*)(kp + 0);
      bk0.q[1] = *(const uint4*)(kp + 16);
      bk1.q[0] = *(const uint4*)(kp + 32);
      bk1.q[1] = *(const uint4*)(kp + 48);
      if (kg + 16 < SS) __builtin_prefetch(kp + (size_t)16 * qs, 0, 1);
    } else {
      uint4 z; z.x = z.y = z.z = z.w = 0u;
      bk0.q[0] = bk0.q[1] = bk1.q[0] = bk1.q[1] = z;
    }
    FragC s;
#pragma unroll
    for (int r = 0; r < 8; ++r) s.f[r] = 0.f;
    s.v = wmma_bf16(aq0.v, bk0.v, s.v);
    s.v = wmma_bf16(aq1.v, bk1.v, s.v);
    int mval = (kg >= 0 && kg < SS) ? amask[tokbase + kg] : 0;
#pragma unroll
    for (int r = 0; r < 8; ++r) {
      int m = lg * 8 + r;
      int dlt = kg - (q0 + m);
      bool valid = (mval > 0) && (dlt >= -WINW) && (dlt <= WINW);
      sc[m][kt * 16 + lm] = valid ? s.f[r] * 0.125f : -1e30f;
    }
  }
  __syncthreads();

  // ---- phase 2: softmax per query row (lanes 0..15) ----
  if (lane < 16) {
    float mx = -1e30f;
    for (int j = 0; j < 528; ++j) mx = fmaxf(mx, sc[lane][j]);
    float sum = 0.f;
    for (int j = 0; j < 528; ++j) {
      float pv = __expf(sc[lane][j] - mx);
      sum += pv;
      pr[lane][j] = f2bf(pv);
    }
    for (int j = 528; j < 544; ++j) pr[lane][j] = 0;
    invs[lane] = 1.0f / sum;
  }
  __syncthreads();

  // ---- phase 3: out = P @ V ----
  FragC o[4];
#pragma unroll
  for (int nt = 0; nt < 4; ++nt)
#pragma unroll
    for (int r = 0; r < 8; ++r) o[nt].f[r] = 0.f;

  for (int c = 0; c < 17; ++c) {
    const int kbase = q0 - WINW + c * 32;
    // batched gather of V chunk to registers (loads pipeline), then transpose
    u32 vv[32];
#pragma unroll
    for (int kk = 0; kk < 32; ++kk) {
      int kg = kbase + kk;
      vv[kk] = 0u;
      if (kg >= 0 && kg < SS)
        vv[kk] = *(const u32*)(vg_ + (tokbase + kg) * qs + hh * DH + lane * 2);
    }
#pragma unroll
    for (int kk = 0; kk < 32; ++kk) {
      vt[lane * 2][kk]     = (u16)(vv[kk] & 0xFFFFu);
      vt[lane * 2 + 1][kk] = (u16)(vv[kk] >> 16);
    }
    __syncthreads();
    FragBF pa;                               // A: m=lm, kb=(lane>>4)*8
    pa.q[0] = *(const uint4*)(&pr[lm][c * 32 + lg * 8]);
    pa.q[1] = *(const uint4*)(&pr[lm][c * 32 + lg * 8 + 16]);
#pragma unroll
    for (int nt = 0; nt < 4; ++nt) {
      FragBF bv;                             // B: n=lm (dh), kb=(lane>>4)*8
      bv.q[0] = *(const uint4*)(&vt[nt * 16 + lm][lg * 8]);
      bv.q[1] = *(const uint4*)(&vt[nt * 16 + lm][lg * 8 + 16]);
      o[nt].v = wmma_bf16(pa.v, bv.v, o[nt].v);
    }
    __syncthreads();
  }

  // ---- phase 4: scale by 1/rowsum, write bf16 [token][h*64+d] ----
#pragma unroll
  for (int nt = 0; nt < 4; ++nt)
#pragma unroll
    for (int r = 0; r < 8; ++r) {
      int m = lg * 8 + r;
      float valf = o[nt].f[r] * invs[m];
      outb[(tokbase + q0 + m) * DD + hh * DH + nt * 16 + lm] = f2bf(valf);
    }
}

// ---------------------------------------------------------------------------
// Final: logits = h @ out_w + out_b ; preds = ALLOWED[argmax]
// ---------------------------------------------------------------------------
__global__ __launch_bounds__(32) void k_logits(const float* __restrict__ h,
                                               const float* __restrict__ ow,
                                               const float* __restrict__ ob,
                                               float* __restrict__ out) {
  __shared__ float slog[NCLS];
  const int t = blockIdx.x, lane = threadIdx.x;
  if (lane < NCLS) {
    const float* hr = h + (size_t)t * DD;
    float acc = ob[lane];
    for (int i = 0; i < DD; ++i) acc += hr[i] * ow[(size_t)i * NCLS + lane];
    out[(size_t)t * NCLS + lane] = acc;
    slog[lane] = acc;
  }
  __syncthreads();
  if (lane == 0) {
    const float allowed[NCLS] = {0, 2, 3, 4, 5, 6, 7, 8, -2, -3, -4, -5, -6, -7, -8};
    int best = 0; float bv = slog[0];
    for (int c = 1; c < NCLS; ++c)
      if (slog[c] > bv) { bv = slog[c]; best = c; }  // first-max tie-break
    out[(size_t)BS * NCLS + t] = allowed[best];
  }
}

// ---------------------------------------------------------------------------
// Host orchestration
// ---------------------------------------------------------------------------
extern "C" void kernel_launch(void* const* d_in, const int* in_sizes, int n_in,
                              void* d_out, int out_size, void* d_ws, size_t ws_size,
                              hipStream_t stream) {
  (void)in_sizes; (void)n_in; (void)out_size; (void)ws_size;

  const float* x       = (const float*)d_in[0];
  const int*   amask   = (const int*)d_in[1];
  const float* in_w    = (const float*)d_in[2];
  const float* in_b    = (const float*)d_in[3];
  const float* pos_emb = (const float*)d_in[4];
  const float* tok_emb = (const float*)d_in[5];
  const float* emb_g   = (const float*)d_in[6];
  const float* emb_b   = (const float*)d_in[7];
  const float* out_w   = (const float*)d_in[8];
  const float* out_b   = (const float*)d_in[9];
  auto LP = [&](int l, int j) -> const float* { return (const float*)d_in[10 + l * 16 + j]; };

  // workspace carve (256B aligned)
  char* wp = (char*)d_ws;
  auto carve = [&](size_t bytes) -> void* {
    void* r = (void*)wp;
    wp += (bytes + 255) & ~(size_t)255;
    return r;
  };
  float* tmp     = (float*)carve((size_t)BS * DD * 4);
  float* h       = (float*)carve((size_t)BS * DD * 4);
  u16*   hbf     = (u16*)carve((size_t)BS * DD * 2);
  u16*   qkvb    = (u16*)carve((size_t)BS * QKVN * 2);
  u16*   ab      = (u16*)carve((size_t)BS * DD * 2);
  u16*   ffb     = (u16*)carve((size_t)BS * FF * 2);
  float* qkvbias = (float*)carve((size_t)QKVN * 4);
  u16*   wT[LLAY][4];
  for (int l = 0; l < LLAY; ++l) {
    wT[l][0] = (u16*)carve((size_t)QKVN * DD * 2);  // fused qkv^T [2304][768]
    wT[l][1] = (u16*)carve((size_t)DD * DD * 2);    // wo^T
    wT[l][2] = (u16*)carve((size_t)FF * DD * 2);    // w1^T [3072][768]
    wT[l][3] = (u16*)carve((size_t)DD * FF * 2);    // w2^T [768][3072]
  }

  // 1) weight transpose + bf16 convert
  auto tlaunch = [&](const float* src, u16* dst, int K, int N) {
    int total = K * N;
    k_transpose_bf16<<<dim3((total + 255) / 256), dim3(256), 0, stream>>>(src, dst, K, N);
  };
  for (int l = 0; l < LLAY; ++l) {
    tlaunch(LP(l, 0),  wT[l][0],                       DD, DD);  // wq -> rows 0..767
    tlaunch(LP(l, 2),  wT[l][0] + (size_t)DD * DD,     DD, DD);  // wk -> rows 768..1535
    tlaunch(LP(l, 4),  wT[l][0] + (size_t)2 * DD * DD, DD, DD);  // wv -> rows 1536..2303
    tlaunch(LP(l, 6),  wT[l][1], DD, DD);                        // wo
    tlaunch(LP(l, 10), wT[l][2], DD, FF);                        // w1
    tlaunch(LP(l, 12), wT[l][3], FF, DD);                        // w2
  }

  // 2) embedding + LN
  k_embed<<<dim3((BS * DD + 255) / 256), dim3(256), 0, stream>>>(x, in_w, in_b, pos_emb,
                                                                 tok_emb, tmp);
  k_ln<<<dim3(BS), dim3(256), 0, stream>>>(tmp, nullptr, emb_g, emb_b, h, hbf);

  // 3) transformer layers
  const dim3 g_qkv(QKVN / 128, BS / 64);   // (18, 64)
  const dim3 g_d(DD / 128, BS / 64);       // (6, 64)
  const dim3 g_ff(FF / 128, BS / 64);      // (24, 64)
  const dim3 g_attn(SS / 16, HH, BB);      // (128, 12, 2)
  for (int l = 0; l < LLAY; ++l) {
    k_concat3<<<dim3((QKVN + 255) / 256), dim3(256), 0, stream>>>(LP(l, 1), LP(l, 3),
                                                                  LP(l, 5), qkvbias);
    k_gemm<<<g_qkv, dim3(128), 0, stream>>>(hbf, wT[l][0], qkvbias, qkvb,
                                            BS, QKVN, DD, 0);
    k_attn<<<g_attn, dim3(32), 0, stream>>>(qkvb, qkvb + DD, qkvb + 2 * DD,
                                            amask, ab, QKVN);
    k_gemm<<<g_d, dim3(128), 0, stream>>>(ab, wT[l][1], LP(l, 7), tmp, BS, DD, DD, 2);
    k_ln<<<dim3(BS), dim3(256), 0, stream>>>(tmp, h, LP(l, 8), LP(l, 9), h, hbf);
    k_gemm<<<g_ff, dim3(128), 0, stream>>>(hbf, wT[l][2], LP(l, 11), ffb, BS, FF, DD, 1);
    k_gemm<<<g_d, dim3(128), 0, stream>>>(ffb, wT[l][3], LP(l, 13), tmp, BS, DD, FF, 2);
    k_ln<<<dim3(BS), dim3(256), 0, stream>>>(tmp, h, LP(l, 14), LP(l, 15), h, hbf);
  }

  // 4) logits + preds
  k_logits<<<dim3(BS), dim3(32), 0, stream>>>(h, out_w, out_b, (float*)d_out);
}
